// RnCLoss_83708912599217
// MI455X (gfx1250) — matrix operand order repaired and checked by reading, not compiled, not run on hardware
//
#include <hip/hip_runtime.h>
#include <hip/hip_bf16.h>
#include <math.h>

// ---------------------------------------------------------------------------
// RnC loss, n = 512 (2*256), D = 512, temperature = 2.
//   G = F F^T via WMMA (f16 Markidis split -> ~fp32 accuracy, f32 accumulate)
//   sq[i,j] = G[i,i] + G[j,j] - 2 G[i,j];  logits = -sqrt(max(sq,0))/2
//   row-max is provably 0 (diagonal logit == 0 exactly) -> skip subtraction
//   denom[i,k] = sum_{j != i, ld_j >= ld_k} exp(logit_j)
//   loss = -(sum_{i,k!=i} (logit_k - log denom_k)) / (n (n-1))
// ---------------------------------------------------------------------------

typedef __attribute__((ext_vector_type(16))) _Float16 v16h;
typedef __attribute__((ext_vector_type(8)))  _Float16 v8h;
typedef __attribute__((ext_vector_type(8)))  float    v8f;

#define NN   512      // n = 2*B
#define DD   512      // feature dim
#define BB   256      // per-branch batch

union V16U { v16h v; v8h h[2]; };

// --- Kernel 1: concat + f16 hi/lo split (Markidis) -------------------------
__global__ __launch_bounds__(256) void rnc_split_f16(
    const float* __restrict__ wt, const float* __restrict__ mt,
    _Float16* __restrict__ fhi, _Float16* __restrict__ flo) {
  int idx = blockIdx.x * blockDim.x + threadIdx.x;   // 0 .. NN*DD-1
  const int BD = BB * DD;
  float v = (idx < BD) ? wt[idx] : mt[idx - BD];
  _Float16 hi = (_Float16)v;
  _Float16 lo = (_Float16)(v - (float)hi);
  fhi[idx] = hi;
  flo[idx] = lo;
}

// --- Kernel 2: Gram matrix G = F F^T via v_wmma_f32_16x16x32_f16 -----------
// One wave (32 lanes) per 16x16 output tile; 32x32 = 1024 tiles.
// A fragment (16x32, MxK), lane layout per ISA 7.12.2:
//   lanes 0-15: M=lane,    VGPR0-3 = K 0..7,  VGPR4-7 = K 16..23
//   lanes 16-31: M=lane-16, VGPR0-3 = K 8..15, VGPR4-7 = K 24..31
// B fragment (32x16, KxN): lanes 0-15 hold N=lane with K 0..7, 16..23 (2/VGPR),
//   lanes 16-31 hold K 8..15, 24..31.  B[k][n] = F[n][k] (G symmetric), so both
//   fragments are contiguous 16B chunks of row-major F rows.
__global__ __launch_bounds__(32) void rnc_gram_wmma(
    const _Float16* __restrict__ fhi, const _Float16* __restrict__ flo,
    float* __restrict__ G) {
  const int tile = blockIdx.x;            // 0..1023
  const int tm = (tile >> 5) * 16;        // output tile row
  const int tn = (tile & 31) * 16;        // output tile col
  const int lane = threadIdx.x;           // 0..31
  const int hl   = lane >> 4;             // half-wave index
  const int l15  = lane & 15;

  const _Float16* aHi = fhi + (size_t)(tm + l15) * DD;
  const _Float16* aLo = flo + (size_t)(tm + l15) * DD;
  const _Float16* bHi = fhi + (size_t)(tn + l15) * DD;
  const _Float16* bLo = flo + (size_t)(tn + l15) * DD;

  v8f acc = {};
  for (int kb = 0; kb < DD; kb += 32) {
    const int a0 = kb + hl * 8;           // A: K group 0..7 / 8..15
    const int a1 = kb + 16 + hl * 8;      // A: K group 16..23 / 24..31
    const int b0 = kb + hl * 8;           // B: K group 0..7 / 8..15 (same as A)

    V16U ahi, alo, bhi, blo;
    ahi.h[0] = *(const v8h*)(aHi + a0);  ahi.h[1] = *(const v8h*)(aHi + a1);
    alo.h[0] = *(const v8h*)(aLo + a0);  alo.h[1] = *(const v8h*)(aLo + a1);
    bhi.h[0] = *(const v8h*)(bHi + b0);  bhi.h[1] = *(const v8h*)(bHi + b0 + 16);
    blo.h[0] = *(const v8h*)(bLo + b0);  blo.h[1] = *(const v8h*)(bLo + b0 + 16);

    // G ~= Fhi*Fhi^T + Fhi*Flo^T + Flo*Fhi^T  (lo*lo term ~2^-22, dropped)
    acc = __builtin_amdgcn_wmma_f32_16x16x32_f16(false, ahi.v, false, bhi.v,
                                                 (short)0, acc, false, false);
    acc = __builtin_amdgcn_wmma_f32_16x16x32_f16(false, ahi.v, false, blo.v,
                                                 (short)0, acc, false, false);
    acc = __builtin_amdgcn_wmma_f32_16x16x32_f16(false, alo.v, false, bhi.v,
                                                 (short)0, acc, false, false);
  }

  // C/D layout: lane<16 -> (M=v, N=lane); lane>=16 -> (M=v+8, N=lane-16)
  const int col   = tn + l15;
  const int rbase = tm + hl * 8;
#pragma unroll
  for (int v = 0; v < 8; ++v)
    G[(size_t)(rbase + v) * NN + col] = acc[v];
}

// --- Kernel 3: per-row denominator pass ------------------------------------
// One block per row i.  LDS holds e_j, ld_j, logit_j for the whole row.
__global__ __launch_bounds__(256) void rnc_row_loss(
    const float* __restrict__ G,
    const float* __restrict__ lwt, const float* __restrict__ lmt,
    float* __restrict__ rowsum) {
  const int i = blockIdx.x;
  const int t = threadIdx.x;

  __shared__ float sE[NN];
  __shared__ float sLd[NN];
  __shared__ float sLogit[NN];
  __shared__ float red[256];

  const float li    = (i < BB) ? lwt[i] : lmt[i - BB];
  const float normi = G[(size_t)i * NN + i];

  for (int j = t; j < NN; j += 256) {
    float lj = (j < BB) ? lwt[j] : lmt[j - BB];
    float nj = G[(size_t)j * NN + j];
    float g  = G[(size_t)i * NN + j];
    float sq = normi + nj - 2.0f * g;
    sq = sq > 0.0f ? sq : 0.0f;
    float dist  = (j == i) ? 0.0f : sqrtf(sq);
    float logit = -0.5f * dist;              // temperature = 2
    sLogit[j] = logit;
    sE[j]     = (j == i) ? 0.0f : expf(logit);  // zero diag -> excluded from denom
    sLd[j]    = fabsf(li - lj);
  }
  __syncthreads();

  float acc = 0.0f;
  for (int k = t; k < NN; k += 256) {
    if (k == i) continue;
    const float ldk = sLd[k];
    float denom = 0.0f;
    for (int j = 0; j < NN; ++j)
      denom += (sLd[j] >= ldk) ? sE[j] : 0.0f;   // sE[i]==0 handles j==i
    acc += sLogit[k] - logf(denom);              // denom >= e_k > 0
  }
  red[t] = acc;
  __syncthreads();
  for (int s = 128; s > 0; s >>= 1) {
    if (t < s) red[t] += red[t + s];
    __syncthreads();
  }
  if (t == 0) rowsum[i] = red[0];
}

// --- Kernel 4: deterministic final reduction -------------------------------
__global__ __launch_bounds__(512) void rnc_final_reduce(
    const float* __restrict__ rowsum, float* __restrict__ out) {
  __shared__ float red[NN];
  const int t = threadIdx.x;
  red[t] = rowsum[t];
  __syncthreads();
  for (int s = 256; s > 0; s >>= 1) {
    if (t < s) red[t] += red[t + s];
    __syncthreads();
  }
  if (t == 0) out[0] = -red[0] / ((float)NN * (float)(NN - 1));
}

// ---------------------------------------------------------------------------
extern "C" void kernel_launch(void* const* d_in, const int* in_sizes, int n_in,
                              void* d_out, int out_size, void* d_ws, size_t ws_size,
                              hipStream_t stream) {
  const float* wt  = (const float*)d_in[0];
  const float* mt  = (const float*)d_in[1];
  const float* lwt = (const float*)d_in[2];
  const float* lmt = (const float*)d_in[3];
  float* out = (float*)d_out;

  // workspace layout (all offsets 16B-aligned): fhi | flo | G | rowsum
  char* ws = (char*)d_ws;
  _Float16* fhi    = (_Float16*)(ws);
  _Float16* flo    = (_Float16*)(ws + (size_t)NN * DD * 2);
  float*    G      = (float*)   (ws + (size_t)NN * DD * 4);
  float*    rowsum = (float*)   (ws + (size_t)NN * DD * 4 + (size_t)NN * NN * 4);

  rnc_split_f16 <<<(NN * DD) / 256, 256, 0, stream>>>(wt, mt, fhi, flo);
  rnc_gram_wmma <<<(NN / 16) * (NN / 16), 32, 0, stream>>>(fhi, flo, G);
  rnc_row_loss  <<<NN, 256, 0, stream>>>(G, lwt, lmt, rowsum);
  rnc_final_reduce<<<1, NN, 0, stream>>>(rowsum, out);
}
